// REN_66082366816535
// MI455X (gfx1250) — matrix operand order repaired and checked by reading, not compiled, not run on hardware
//
#include <hip/hip_runtime.h>
#include <hip/hip_bf16.h>
#include <math.h>

#define D_MODEL   1024
#define NBLK      20
#define NB_PAD    32
#define BATCH     8
#define SEQ       2048
#define HT_STRIDE 36      // bf16 elems per h^T row: 72B rows (8B aligned), 18-bank step
#define HM_STRIDE 1025    // f32 elems per h row: odd stride -> conflict-free column access
#define EPS_N     1e-8f

#define WX_TILES  ((BATCH * SEQ / 16) * (D_MODEL / 16))   // 65536 output tiles
#define WX_BLOCKS (WX_TILES / 8)                          // 8 waves per block -> 8192

typedef __bf16 bf16_t;
typedef __attribute__((ext_vector_type(4)))  __bf16 v4bf;
typedef __attribute__((ext_vector_type(8)))  __bf16 v8bf;
typedef __attribute__((ext_vector_type(16))) __bf16 v16bf;
typedef __attribute__((ext_vector_type(8)))  float  v8f;

static __device__ __forceinline__ v16bf concat8(v8bf lo, v8bf hi) {
  return __builtin_shufflevector(lo, hi, 0,1,2,3,4,5,6,7,8,9,10,11,12,13,14,15);
}

// B fragment (32x16 bf16): lane = K-row, 16 contiguous N values per lane (8B-aligned LDS)
static __device__ __forceinline__ v16bf load_b16_lds(const bf16_t* p) {
  const v4bf* q = (const v4bf*)p;
  v4bf a = q[0], b = q[1], c = q[2], d = q[3];
  v8bf lo = __builtin_shufflevector(a, b, 0,1,2,3,4,5,6,7);
  v8bf hi = __builtin_shufflevector(c, d, 0,1,2,3,4,5,6,7);
  return concat8(lo, hi);
}

static __device__ __forceinline__ v8f wmma_bf16(v16bf a, v16bf b, v8f c) {
  // D = A x B + C ; f32 accumulate
  return __builtin_amdgcn_wmma_f32_16x16x32_bf16(false, a, false, b, (short)0, c, false, false);
}

// ---------------- prep kernels (parallel, run once, whole GPU) ----------------

__global__ void k_cvt_bf16(const float* __restrict__ s, bf16_t* __restrict__ d, int n) {
  int i = blockIdx.x * 256 + threadIdx.x;
  if (i < n) d[i] = (bf16_t)s[i];
}

__global__ void k_trans_W(const float* __restrict__ W, bf16_t* __restrict__ Wtb) {
  int i = blockIdx.x * 256 + threadIdx.x;         // i = n*1024 + k (row n of W, no transpose)
  int k = i >> 10, n = i & 1023;
  Wtb[i] = (bf16_t)W[k * D_MODEL + n];
}

__global__ void k_xsum(const float* __restrict__ e, const float* __restrict__ m,
                       bf16_t* __restrict__ xb, int n) {
  int i = blockIdx.x * 256 + threadIdx.x;
  if (i < n) xb[i] = (bf16_t)(e[i] + m[i]);
}

// kvT[d*20+nb] = (keys @ V^T)[nb][d] = sum_k keys[nb,k] * V[d,k]   (constant!)
__global__ void k_kvT(const float* __restrict__ keys, const float* __restrict__ V,
                      float* __restrict__ kvT) {
  int i = blockIdx.x * 256 + threadIdx.x;
  if (i >= D_MODEL * NBLK) return;
  int d = i / NBLK, nb = i % NBLK;
  float acc = 0.f;
  for (int k = 0; k < D_MODEL; ++k) acc += keys[nb * D_MODEL + k] * V[d * D_MODEL + k];
  kvT[i] = acc;
}

// kx[bt*20+nb] = keys @ x_t  for all timesteps
__global__ void k_kx(const bf16_t* __restrict__ xb, const float* __restrict__ keys,
                     float* __restrict__ kx) {
  int i = blockIdx.x * 256 + threadIdx.x;
  if (i >= BATCH * SEQ * NBLK) return;
  long long bt = i / NBLK; int nb = i % NBLK;
  float acc = 0.f;
  for (int k = 0; k < D_MODEL; ++k)
    acc += (float)xb[bt * D_MODEL + k] * keys[nb * D_MODEL + k];
  kx[i] = acc;
}

// Wx = x @ W^T for all (b,t): bf16 WMMA, M=16384 N=1024 K=1024. One 16x16 tile per wave.
__global__ __launch_bounds__(256) void k_wx(const bf16_t* __restrict__ xb,
                                            const bf16_t* __restrict__ Wtb,
                                            float* __restrict__ Wx) {
  const int lane = threadIdx.x & 31;
  const int wv   = threadIdx.x >> 5;
  const int tile = blockIdx.x * 8 + wv;
  if (tile >= WX_TILES) return;                   // wave-uniform guard
  const int mt = tile >> 6, nt = tile & 63;
  const int laneN = lane & 15, half = lane >> 4;
  const size_t mrow = (size_t)(mt * 16 + laneN);

  v8f c = {};
  for (int k0 = 0; k0 < D_MODEL; k0 += 32) {
    const int kb = k0 + 8 * half;
    v16bf A = concat8(*(const v8bf*)(xb + mrow * D_MODEL + kb),
                      *(const v8bf*)(xb + mrow * D_MODEL + kb + 16));
    const size_t nrow = (size_t)(nt * 16 + laneN);
    v16bf B = concat8(*(const v8bf*)(Wtb + nrow * D_MODEL + kb),
                      *(const v8bf*)(Wtb + nrow * D_MODEL + kb + 16));
    c = wmma_bf16(A, B, c);
  }
  #pragma unroll
  for (int r = 0; r < 8; ++r) {
    size_t row = (size_t)mt * 16 + r + 8 * half;
    Wx[row * D_MODEL + nt * 16 + laneN] = c[r];
  }
}

// ---------------- sequential recurrent kernel: one workgroup per batch ----------------

__global__ __launch_bounds__(1024, 1) void k_rnn(
    const float*  __restrict__ state,   // [NBLK*D]
    const bf16_t* __restrict__ Ub,      // [D][D] row-major bf16
    const bf16_t* __restrict__ xb,      // [B*S][D] bf16
    const float*  __restrict__ Wxg,     // [B*S][D]
    const float*  __restrict__ kvTg,    // [D][NBLK]
    const float*  __restrict__ kxg,     // [B*S][NBLK]
    float*        __restrict__ out)     // [B][NBLK][D]
{
  __shared__ float  hm_s[NBLK * HM_STRIDE];        // f32 master h   (82 KB)
  __shared__ bf16_t hT_s[D_MODEL * HT_STRIDE];     // bf16 h^T       (72 KB)
  __shared__ float  wx_s[D_MODEL];
  __shared__ float  g_s[NB_PAD];
  __shared__ float  n2_s[NB_PAD];

  const int tid  = threadIdx.x;
  const int lane = tid & 31;
  const int wv   = tid >> 5;
  const int b    = blockIdx.x;
  const int dcol = tid;

  // ---- init: h0 shared across batches; h^T padding columns stay zero forever
  #pragma unroll
  for (int j = 0; j < HT_STRIDE; ++j) hT_s[dcol * HT_STRIDE + j] = (bf16_t)0.0f;
  for (int nb = 0; nb < NBLK; ++nb) {
    float v = state[nb * D_MODEL + dcol];
    hm_s[nb * HM_STRIDE + dcol] = v;
    hT_s[dcol * HT_STRIDE + nb] = (bf16_t)v;
  }
  __syncthreads();

  const int laneN = lane & 15;
  const int half  = lane >> 4;
  const size_t m0row = (size_t)(2 * wv) * 16 + laneN;   // U row, M-tile 2w
  const size_t m1row = m0row + 16;                      // U row, M-tile 2w+1

  for (int t = 0; t < SEQ; ++t) {
    const long long bt = (long long)b * SEQ + t;

    // phase 0: stage Wx row; clear norm accumulators; prefetch next step's rows
    wx_s[dcol] = Wxg[bt * D_MODEL + dcol];
    if (tid < NB_PAD) n2_s[tid] = 0.0f;
    if (t + 1 < SEQ) {
      __builtin_prefetch(Wxg + (bt + 1) * D_MODEL + dcol, 0, 1);  // global_prefetch_b8
      __builtin_prefetch(xb  + (bt + 1) * D_MODEL + dcol, 0, 1);
    }
    __syncthreads();

    // gates: wave nb computes h[nb]·x_t (f32 master h, bf16 x) -> sigmoid(+keys·x)
    if (wv < NBLK) {
      const bf16_t* xrow = xb + bt * D_MODEL;
      float acc = 0.0f;
      #pragma unroll 4
      for (int i = 0; i < D_MODEL / 32; ++i) {
        int d = lane + 32 * i;
        acc += hm_s[wv * HM_STRIDE + d] * (float)xrow[d];
      }
      #pragma unroll
      for (int off = 16; off > 0; off >>= 1) acc += __shfl_xor(acc, off, 32);
      if (lane == 0) g_s[wv] = 1.0f / (1.0f + expf(-(acc + kxg[bt * NBLK + wv])));
    }

    // WMMA: candT = U @ h^T   (M=1024, N=32 pad, K=1024); wave owns m in [32w,32w+32)
    v8f c00 = {}, c01 = {}, c10 = {}, c11 = {};
    #pragma unroll 2
    for (int k0 = 0; k0 < D_MODEL; k0 += 32) {
      const int kb = k0 + 8 * half;
      v16bf A0 = concat8(*(const v8bf*)(Ub + m0row * D_MODEL + kb),
                         *(const v8bf*)(Ub + m0row * D_MODEL + kb + 16));
      v16bf A1 = concat8(*(const v8bf*)(Ub + m1row * D_MODEL + kb),
                         *(const v8bf*)(Ub + m1row * D_MODEL + kb + 16));
      const bf16_t* hcol = &hT_s[k0 * HT_STRIDE + laneN];
      v16bf B0, B1;
      #pragma unroll
      for (int s = 0; s < 16; ++s) {
        const int kk = (s < 8) ? (8 * half + s) : (16 + 8 * half + (s - 8));
        B0[s] = hcol[kk * HT_STRIDE];        // slots 0..15
        B1[s] = hcol[kk * HT_STRIDE + 16];   // slots 16..31 (20..31 are zero pad)
      }
      c00 = wmma_bf16(A0, B0, c00);
      c01 = wmma_bf16(A0, B1, c01);
      c10 = wmma_bf16(A1, B0, c10);
      c11 = wmma_bf16(A1, B1, c11);
    }
    __syncthreads();   // gates visible; gate-phase hm reads complete before updates

    // update: h += g * tanh(candT + kvT + Wx), accumulate per-slot sum of squares
    {
      const float gA = g_s[laneN];
      const float gB = (laneN < 4) ? g_s[16 + laneN] : 0.0f;
      float ss0 = 0.0f, ss1 = 0.0f;
      const int mb0 = (2 * wv) * 16 + 8 * half;
      #pragma unroll
      for (int r = 0; r < 8; ++r) {
        int m = mb0 + r;
        float z  = c00[r] + kvTg[m * NBLK + laneN] + wx_s[m];
        float hv = hm_s[laneN * HM_STRIDE + m] + gA * tanhf(z);
        hm_s[laneN * HM_STRIDE + m] = hv;  ss0 += hv * hv;
        if (laneN < 4) {
          int nb = 16 + laneN;
          float z1  = c01[r] + kvTg[m * NBLK + nb] + wx_s[m];
          float hv1 = hm_s[nb * HM_STRIDE + m] + gB * tanhf(z1);
          hm_s[nb * HM_STRIDE + m] = hv1;  ss1 += hv1 * hv1;
        }
      }
      const int mb1 = mb0 + 16;
      #pragma unroll
      for (int r = 0; r < 8; ++r) {
        int m = mb1 + r;
        float z  = c10[r] + kvTg[m * NBLK + laneN] + wx_s[m];
        float hv = hm_s[laneN * HM_STRIDE + m] + gA * tanhf(z);
        hm_s[laneN * HM_STRIDE + m] = hv;  ss0 += hv * hv;
        if (laneN < 4) {
          int nb = 16 + laneN;
          float z1  = c11[r] + kvTg[m * NBLK + nb] + wx_s[m];
          float hv1 = hm_s[nb * HM_STRIDE + m] + gB * tanhf(z1);
          hm_s[nb * HM_STRIDE + m] = hv1;  ss1 += hv1 * hv1;
        }
      }
      ss0 += __shfl_xor(ss0, 16, 32);
      ss1 += __shfl_xor(ss1, 16, 32);
      if (lane < 16) {
        atomicAdd(&n2_s[laneN], ss0);                       // ds_add_f32
        if (laneN < 4) atomicAdd(&n2_s[16 + laneN], ss1);
      }
    }
    __syncthreads();

    // normalize: h /= (||h|| + eps); refresh bf16 h^T for next step's WMMA
    #pragma unroll 4
    for (int nb = 0; nb < NBLK; ++nb) {
      float inv = 1.0f / (sqrtf(n2_s[nb]) + EPS_N);
      float v = hm_s[nb * HM_STRIDE + dcol] * inv;
      hm_s[nb * HM_STRIDE + dcol] = v;
      hT_s[dcol * HT_STRIDE + nb] = (bf16_t)v;
    }
    __syncthreads();
  }

  for (int nb = 0; nb < NBLK; ++nb)
    out[((size_t)b * NBLK + nb) * D_MODEL + dcol] = hm_s[nb * HM_STRIDE + dcol];
}

// ---------------- launch ----------------

extern "C" void kernel_launch(void* const* d_in, const int* in_sizes, int n_in,
                              void* d_out, int out_size, void* d_ws, size_t ws_size,
                              hipStream_t stream) {
  const float* e     = (const float*)d_in[0];
  const float* m     = (const float*)d_in[1];
  const float* state = (const float*)d_in[2];
  const float* U     = (const float*)d_in[3];
  const float* V     = (const float*)d_in[4];
  const float* W     = (const float*)d_in[5];
  const float* keys  = (const float*)d_in[6];
  float* out = (float*)d_out;

  char* ws = (char*)d_ws;
  const size_t DD   = (size_t)D_MODEL * D_MODEL;      // 1M elems
  const size_t BSD  = (size_t)BATCH * SEQ * D_MODEL;  // 16.7M elems
  bf16_t* Ub  = (bf16_t*)(ws);                            // 2 MB
  bf16_t* Wtb = (bf16_t*)(ws + DD * 2);                   // 2 MB
  bf16_t* xb  = (bf16_t*)(ws + DD * 4);                   // 32 MB
  float*  Wx  = (float*) (ws + DD * 4 + BSD * 2);         // 64 MB
  float*  kvT = (float*) (ws + DD * 4 + BSD * 6);         // 80 KB
  float*  kx  = (float*) (ws + DD * 4 + BSD * 6 + (size_t)D_MODEL * NBLK * 4);

  // prep (fully parallel)
  k_cvt_bf16<<<(int)(DD / 256), 256, 0, stream>>>(U, Ub, (int)DD);
  k_trans_W <<<(int)(DD / 256), 256, 0, stream>>>(W, Wtb);
  k_xsum    <<<(int)(BSD / 256), 256, 0, stream>>>(e, m, xb, (int)BSD);
  k_kvT     <<<(D_MODEL * NBLK + 255) / 256, 256, 0, stream>>>(keys, V, kvT);
  k_kx      <<<(BATCH * SEQ * NBLK + 255) / 256, 256, 0, stream>>>(xb, keys, kx);
  k_wx      <<<WX_BLOCKS, 256, 0, stream>>>(xb, Wtb, Wx);   // 8192 blocks, 65536 tiles

  // sequential recurrence: one WGP-filling workgroup per batch chain
  k_rnn<<<BATCH, 1024, 0, stream>>>(state, Ub, xb, Wx, kvT, kx, out);
}
